// DualRoadGNN_24077586661864
// MI455X (gfx1250) — matrix operand order repaired and checked by reading, not compiled, not run on hardware
//
#include <hip/hip_runtime.h>
#include <hip/hip_bf16.h>
#include <cmath>

// ---------------- problem constants (match reference) ----------------
#define Gc    100
#define NPGc  500
#define Nn    (Gc * NPGc)      // 50000
#define INc   128
#define Hc    256
#define Lc    3
#define KNN_K 3
#define STRIPS 32              // ceil(NPGc/16)

typedef __attribute__((ext_vector_type(16))) _Float16 v16h;
typedef __attribute__((ext_vector_type(8)))  float    v8f;
typedef __attribute__((ext_vector_type(4)))  float    f32x4;

union H16 { v16h v; f32x4 f4[2]; };

// ---------------- conversion kernels ----------------
__global__ __launch_bounds__(256) void cvt_f16_kernel(const float* __restrict__ in,
                                                      _Float16* __restrict__ out, int n) {
    int i = blockIdx.x * 256 + threadIdx.x;
    if (i < n) out[i] = (_Float16)in[i];
}

// normalized f16: out[n,d] = (f16)(h[n,d] * rnorm[n])
__global__ __launch_bounds__(256)
void cvt_scale_f16_kernel(const float* __restrict__ h, const float* __restrict__ rnorm,
                          _Float16* __restrict__ out, int n) {
    int i = blockIdx.x * 256 + threadIdx.x;
    if (i < n) out[i] = (_Float16)(h[i] * rnorm[i >> 8]);   // Hc == 256
}

// in: [K, Nc] row-major f32  ->  out: [Nc, K] row-major f16 (transposed)
__global__ __launch_bounds__(256) void cvtT_f16_kernel(const float* __restrict__ in,
                                                       _Float16* __restrict__ out,
                                                       int Kd, int Nc) {
    int i = blockIdx.x * 256 + threadIdx.x;
    if (i < Kd * Nc) {
        int k = i / Nc, n = i % Nc;
        out[(size_t)n * Kd + k] = (_Float16)in[i];
    }
}

// ---------------- WMMA GEMM: C[M,Nc] = A[M,K] * W[K,Nc] (+bias) ----------------
// A f16 row-major (optionally split at k==ksplit between A0/A1).
// W transposed f16: WT[Nc,K]. One wave -> one 16x16 M-tile x four N-tiles.
__global__ __launch_bounds__(256)
void wmma_gemm_kernel(const _Float16* __restrict__ A0,
                      const _Float16* __restrict__ A1, int ksplit,
                      const _Float16* __restrict__ WT,
                      const float* __restrict__ bias,
                      float* __restrict__ C,
                      int M, int Kd, int Nc) {
    int wave = threadIdx.x >> 5;
    int lane = threadIdx.x & 31;
    int tm  = blockIdx.x * 8 + wave;          // M-tile
    int tnB = blockIdx.y * 4;                 // first of 4 N-tiles
    if (tm * 16 >= M) return;                 // uniform per wave: EXEC stays all-1s

    int hi    = lane >> 4;                    // lane half
    int ln    = lane & 15;
    int koffA = hi * 8;                       // ISA A-fragment layout (16x32 f16)
    int koffB = hi * 8;                       // ISA B-fragment layout (32x16 f16)
    int rowA  = tm * 16 + ln;

    v8f acc[4] = {{}, {}, {}, {}};
    for (int kk = 0; kk < Kd; kk += 32) {
        const _Float16* aptr;
        if (kk < ksplit) aptr = A0 + (size_t)rowA * ksplit + kk + koffA;
        else             aptr = A1 + (size_t)rowA * (Kd - ksplit) + (kk - ksplit) + koffA;
        __builtin_prefetch(aptr + 128, 0, 1);                 // global_prefetch_b8
        H16 ua;
        ua.f4[0] = *(const f32x4*)(aptr);                     // k = base..base+7
        ua.f4[1] = *(const f32x4*)(aptr + 16);                // k = base+16..base+23
#pragma unroll
        for (int j = 0; j < 4; ++j) {
            const _Float16* bptr = WT + (size_t)((tnB + j) * 16 + ln) * Kd + kk + koffB;
            H16 ub;
            ub.f4[0] = *(const f32x4*)(bptr);
            ub.f4[1] = *(const f32x4*)(bptr + 16);
            acc[j] = __builtin_amdgcn_wmma_f32_16x16x32_f16(false, ua.v, false, ub.v,
                                                            (short)0, acc[j], false, false);
        }
    }
    int rbase = tm * 16 + hi * 8;             // ISA C/D layout: VGPR r -> M = r + hi*8
#pragma unroll
    for (int j = 0; j < 4; ++j) {
        int col  = (tnB + j) * 16 + ln;
        float bv = bias ? bias[col] : 0.0f;
#pragma unroll
        for (int r = 0; r < 8; ++r)
            C[(size_t)(rbase + r) * Nc + col] = acc[j][r] + bv;
    }
}

// ---------------- reciprocal norm per node: 1/(||h||+1e-12) ----------------
__global__ __launch_bounds__(256)
void rnorm_kernel(const float* __restrict__ h, float* __restrict__ rnorm, int n) {
    int wave = threadIdx.x >> 5, lane = threadIdx.x & 31;
    int node = blockIdx.x * 8 + wave;
    if (node >= n) return;
    const float* p = h + (size_t)node * Hc;
    float ss = 0.f;
#pragma unroll
    for (int j = 0; j < Hc / 32; ++j) { float v = p[lane + j * 32]; ss += v * v; }
#pragma unroll
    for (int off = 16; off > 0; off >>= 1) ss += __shfl_xor(ss, off, 32);
    if (lane == 0) rnorm[node] = 1.0f / (sqrtf(ss) + 1e-12f);
}

// ---------------- WMMA cosine kNN ----------------
// hn: [Nn, Hc] f16 row-normalized features. One wave handles a 16-dst strip of one
// graph; loops over 32 candidate tiles; sim tile = WMMA(A=dst rows, B=cand rows);
// lane l (<16) owns dst row l and keeps a running top-3 (ascending candidate order
// preserves jax top_k's lowest-index tie-break).
__global__ __launch_bounds__(32)
void knn_wmma_kernel(const _Float16* __restrict__ hn, int* __restrict__ fsrc) {
    __shared__ float sim[16][17];
    int strip = blockIdx.x;
    int g = strip >> 5, s = strip & (STRIPS - 1);
    int gBase = g * NPGc;
    int lane = threadIdx.x;
    int hi = lane >> 4, ln = lane & 15;
    int koffA = hi * 8, koffB = hi * 8;

    // hoist all 8 A fragments (16 dst rows x 256 dims), clamped to valid rows
    int rIn = s * 16 + ln;
    int rClamp = rIn < NPGc ? rIn : NPGc - 1;
    const _Float16* aBase = hn + (size_t)(gBase + rClamp) * Hc + koffA;
    v16h aF[8];
#pragma unroll
    for (int k8 = 0; k8 < 8; ++k8) {
        H16 ua;
        ua.f4[0] = *(const f32x4*)(aBase + k8 * 32);
        ua.f4[1] = *(const f32x4*)(aBase + k8 * 32 + 16);
        aF[k8] = ua.v;
    }

    float bv[KNN_K] = {-1e30f, -1e30f, -1e30f};
    int   bi[KNN_K] = {0x7fffffff, 0x7fffffff, 0x7fffffff};

    for (int ct = 0; ct < STRIPS; ++ct) {
        int cIn = ct * 16 + ln;
        int cClamp = cIn < NPGc ? cIn : NPGc - 1;
        const _Float16* bBase = hn + (size_t)(gBase + cClamp) * Hc + koffB;
        v8f acc = {};
#pragma unroll
        for (int k8 = 0; k8 < 8; ++k8) {
            H16 ub;
            ub.f4[0] = *(const f32x4*)(bBase + k8 * 32);
            ub.f4[1] = *(const f32x4*)(bBase + k8 * 32 + 16);
            acc = __builtin_amdgcn_wmma_f32_16x16x32_f16(false, aF[k8], false, ub.v,
                                                         (short)0, acc, false, false);
        }
        __syncthreads();                       // previous scan done before overwrite
#pragma unroll
        for (int r = 0; r < 8; ++r) sim[r + hi * 8][ln] = acc[r];
        __syncthreads();
        if (lane < 16) {
#pragma unroll
            for (int n = 0; n < 16; ++n) {
                int c = ct * 16 + n;
                if (c < NPGc) {
                    float sv = sim[lane][n];
                    int idx = gBase + c;
                    if (sv > bv[0]) { bv[2]=bv[1]; bi[2]=bi[1]; bv[1]=bv[0]; bi[1]=bi[0]; bv[0]=sv; bi[0]=idx; }
                    else if (sv > bv[1]) { bv[2]=bv[1]; bi[2]=bi[1]; bv[1]=sv; bi[1]=idx; }
                    else if (sv > bv[2]) { bv[2]=sv; bi[2]=idx; }
                }
            }
        }
    }
    if (lane < 16 && (s * 16 + lane) < NPGc) {
        int t = gBase + s * 16 + lane;
#pragma unroll
        for (int kp = 0; kp < KNN_K; ++kp) fsrc[t * KNN_K + kp] = bi[kp];
    }
}

// ---------------- degree / dinv ----------------
__global__ void fill_int_kernel(int* p, int n, int v) {
    int i = blockIdx.x * 256 + threadIdx.x; if (i < n) p[i] = v;
}
__global__ void fill_float_kernel(float* p, int n, float v) {
    int i = blockIdx.x * 256 + threadIdx.x; if (i < n) p[i] = v;
}
__global__ void edge_count_kernel(const int* __restrict__ dst, int* __restrict__ deg, int E) {
    int e = blockIdx.x * 256 + threadIdx.x; if (e < E) atomicAdd(&deg[dst[e]], 1);
}
__global__ void dinv_kernel(const int* __restrict__ deg, float* __restrict__ dinv, int n) {
    int i = blockIdx.x * 256 + threadIdx.x; if (i < n) dinv[i] = rsqrtf((float)deg[i]);
}

// ---------------- GCN aggregation ----------------
__global__ __launch_bounds__(256)
void init_self_kernel(float* __restrict__ out, const float* __restrict__ m,
                      const float* __restrict__ bias, const float* __restrict__ dinv) {
    int n = blockIdx.x, c = threadIdx.x;
    float d = dinv[n];
    out[(size_t)n * Hc + c] = bias[c] + m[(size_t)n * Hc + c] * d * d;
}
__global__ __launch_bounds__(256)
void edge_scatter_kernel(const float* __restrict__ m, const int* __restrict__ src,
                         const int* __restrict__ dst, const float* __restrict__ dinv,
                         float* __restrict__ out) {
    int e = blockIdx.x, c = threadIdx.x;
    int s = src[e], d = dst[e];
    float coef = dinv[s] * dinv[d];
    atomicAdd(&out[(size_t)d * Hc + c], m[(size_t)s * Hc + c] * coef);
}
__global__ __launch_bounds__(256)
void knn_gather_kernel(const float* __restrict__ m, const int* __restrict__ fsrc,
                       const float* __restrict__ dinv, const float* __restrict__ bias,
                       float* __restrict__ out) {
    int n = blockIdx.x, c = threadIdx.x;
    float dn = dinv[n];
    float acc = bias[c] + m[(size_t)n * Hc + c] * dn * dn;
#pragma unroll
    for (int k = 0; k < KNN_K; ++k) {
        int s = fsrc[n * KNN_K + k];
        acc += m[(size_t)s * Hc + c] * dinv[s] * dn;
    }
    out[(size_t)n * Hc + c] = acc;
}

// ---------------- fused GraphNorm + leaky_relu (in-place), block = one graph ----------------
__global__ __launch_bounds__(256)
void graphnorm_act_kernel(float* __restrict__ buf, const float* __restrict__ w,
                          const float* __restrict__ b, const float* __restrict__ ms) {
    int g = blockIdx.x, t = threadIdx.x;     // t == feature column (Hc == blockDim)
    float* p = buf + (size_t)g * NPGc * Hc + t;
    float s = 0.f;
    for (int n = 0; n < NPGc; ++n) s += p[(size_t)n * Hc];
    float mean = s / (float)NPGc;
    float sub = mean * ms[t];
    float v = 0.f;
    for (int n = 0; n < NPGc; ++n) { float o = p[(size_t)n * Hc] - sub; v += o * o; }
    float rstd = rsqrtf(v / (float)NPGc + 1e-5f);
    float wt = w[t], bt = b[t];
    for (int n = 0; n < NPGc; ++n) {
        float o = (p[(size_t)n * Hc] - sub) * rstd * wt + bt;
        p[(size_t)n * Hc] = o > 0.f ? o : 0.01f * o;     // leaky_relu(0.01)
    }
}

// ---------------- gate + residual: h = g*hA + (1-g)*fB + h_prev ----------------
__global__ __launch_bounds__(256)
void gate_finalize_kernel(float* __restrict__ h, const float* __restrict__ hA,
                          const float* __restrict__ fB, const float* __restrict__ logits,
                          int n) {
    int i = blockIdx.x * 256 + threadIdx.x;
    if (i < n) {
        float gt = 1.0f / (1.0f + expf(-logits[i]));
        h[i] = gt * hA[i] + (1.0f - gt) * fB[i] + h[i];
    }
}

// ---------------- mean-pool per graph, out (+)= wgt * mean ----------------
__global__ __launch_bounds__(256)
void pool_kernel(const float* __restrict__ h, float* __restrict__ out,
                 float wgt, int accumulate) {
    int g = blockIdx.x, t = threadIdx.x;
    const float* p = h + (size_t)g * NPGc * Hc + t;
    float s = 0.f;
    for (int n = 0; n < NPGc; ++n) s += p[(size_t)n * Hc];
    float v = wgt * s / (float)NPGc;
    if (accumulate) out[g * Hc + t] += v;
    else            out[g * Hc + t] = v;
}

// ---------------- host orchestration ----------------
extern "C" void kernel_launch(void* const* d_in, const int* in_sizes, int n_in,
                              void* d_out, int out_size, void* d_ws, size_t ws_size,
                              hipStream_t stream) {
    const float* x       = (const float*)d_in[0];
    const int*   ei      = (const int*)d_in[1];
    const float* emb_W   = (const float*)d_in[3];
    const float* emb_b   = (const float*)d_in[4];
    const float* conv_W  = (const float*)d_in[5];
    const float* conv_b  = (const float*)d_in[6];
    const float* fconv_W = (const float*)d_in[7];
    const float* fconv_b = (const float*)d_in[8];
    const float* norm_w  = (const float*)d_in[9];
    const float* norm_b  = (const float*)d_in[10];
    const float* norm_ms = (const float*)d_in[11];
    const float* fnorm_w = (const float*)d_in[12];
    const float* fnorm_b = (const float*)d_in[13];
    const float* fnorm_ms= (const float*)d_in[14];
    const float* gate_W  = (const float*)d_in[15];
    const float* gate_b  = (const float*)d_in[16];
    const int E = in_sizes[1] / 2;
    const int* esrc = ei;
    const int* edst = ei + E;
    float* out = (float*)d_out;

    // ---- workspace carve (all 16B-aligned) ----
    const size_t NH = (size_t)Nn * Hc;          // 12.8M
    float* h   = (float*)d_ws;
    float* hA  = h  + NH;
    float* fB  = hA + NH;
    float* m   = fB + NH;                       // GEMM output / gate logits
    _Float16* a16 = (_Float16*)(m + NH);        // also holds x16 (N*IN <= N*H)
    _Float16* b16 = a16 + NH;                   // hn16 for knn, later fB f16
    _Float16* embT   = b16 + NH;                        // [Hc, INc]
    _Float16* convT0 = embT   + (size_t)Hc * INc;       // [Hc, Hc]
    _Float16* convT1 = convT0 + (size_t)Hc * Hc;
    _Float16* fconvT0= convT1 + (size_t)Hc * Hc;
    _Float16* fconvT1= fconvT0 + (size_t)Hc * Hc;
    _Float16* gateT  = fconvT1 + (size_t)Hc * Hc;       // [Hc, 2*Hc]
    float* rnorm  = (float*)(gateT + (size_t)Hc * 2 * Hc);
    float* dinv_e = rnorm  + Nn;
    float* dinv_f = dinv_e + Nn;
    int*   degI   = (int*)(dinv_f + Nn);
    int*   fsrc   = degI + Nn;

    auto blocks = [](size_t n) { return (unsigned)((n + 255) / 256); };
    auto gemm = [&](const _Float16* A0, const _Float16* A1, int ksplit,
                    const _Float16* WT, const float* bias, float* C, int Kd) {
        dim3 grid((Nn / 16 + 7) / 8, Hc / 64);
        wmma_gemm_kernel<<<grid, 256, 0, stream>>>(A0, A1, ksplit, WT, bias, C, Nn, Kd, Hc);
    };

    // ---- weight conversion (transposed f16) ----
    cvtT_f16_kernel<<<blocks((size_t)INc * Hc), 256, 0, stream>>>(emb_W, embT, INc, Hc);
    cvtT_f16_kernel<<<blocks((size_t)Hc * Hc), 256, 0, stream>>>(conv_W,          convT0, Hc, Hc);
    cvtT_f16_kernel<<<blocks((size_t)Hc * Hc), 256, 0, stream>>>(conv_W + Hc*Hc,  convT1, Hc, Hc);
    cvtT_f16_kernel<<<blocks((size_t)Hc * Hc), 256, 0, stream>>>(fconv_W,         fconvT0, Hc, Hc);
    cvtT_f16_kernel<<<blocks((size_t)Hc * Hc), 256, 0, stream>>>(fconv_W + Hc*Hc, fconvT1, Hc, Hc);
    cvtT_f16_kernel<<<blocks((size_t)2 * Hc * Hc), 256, 0, stream>>>(gate_W, gateT, 2 * Hc, Hc);

    // ---- embedding: h = x @ emb_W + emb_b ----
    cvt_f16_kernel<<<blocks((size_t)Nn * INc), 256, 0, stream>>>(x, a16, Nn * INc);
    gemm(a16, a16, INc, embT, emb_b, h, INc);

    // ---- kNN graph on embedded features (WMMA similarity) ----
    rnorm_kernel<<<(Nn + 7) / 8, 256, 0, stream>>>(h, rnorm, Nn);
    cvt_scale_f16_kernel<<<blocks(NH), 256, 0, stream>>>(h, rnorm, b16, (int)NH);
    knn_wmma_kernel<<<Gc * STRIPS, 32, 0, stream>>>(b16, fsrc);

    // ---- degrees: edge graph (self loop + counts), kNN graph (constant K+1) ----
    fill_int_kernel<<<blocks(Nn), 256, 0, stream>>>(degI, Nn, 1);
    edge_count_kernel<<<blocks(E), 256, 0, stream>>>(edst, degI, E);
    dinv_kernel<<<blocks(Nn), 256, 0, stream>>>(degI, dinv_e, Nn);
    fill_float_kernel<<<blocks(Nn), 256, 0, stream>>>(dinv_f, Nn, 0.5f);  // rsqrt(K+1)=rsqrt(4)

    // ---- layers (L-1 = 2 iterations, unrolled on host) ----
    const _Float16* cT[2] = {convT0, convT1};
    const _Float16* fT[2] = {fconvT0, fconvT1};
    for (int i = 0; i < Lc - 1; ++i) {
        // structural road: hA = leaky(norm(gcn(h)))
        cvt_f16_kernel<<<blocks(NH), 256, 0, stream>>>(h, a16, (int)NH);
        gemm(a16, a16, Hc, cT[i], nullptr, m, Hc);
        init_self_kernel<<<Nn, Hc, 0, stream>>>(hA, m, conv_b + i * Hc, dinv_e);
        edge_scatter_kernel<<<E, Hc, 0, stream>>>(m, esrc, edst, dinv_e, hA);
        graphnorm_act_kernel<<<Gc, Hc, 0, stream>>>(hA, norm_w + i * Hc, norm_b + i * Hc,
                                                    norm_ms + i * Hc);
        // feature road: fB = leaky(norm(gcn_knn(hA)))
        cvt_f16_kernel<<<blocks(NH), 256, 0, stream>>>(hA, a16, (int)NH);
        gemm(a16, a16, Hc, fT[i], nullptr, m, Hc);
        knn_gather_kernel<<<Nn, Hc, 0, stream>>>(m, fsrc, dinv_f, fconv_b + i * Hc, fB);
        graphnorm_act_kernel<<<Gc, Hc, 0, stream>>>(fB, fnorm_w + i * Hc, fnorm_b + i * Hc,
                                                    fnorm_ms + i * Hc);
        // gate: sigmoid(concat(hA,fB) @ gate_W + gate_b); h = g*hA + (1-g)*fB + h_prev
        cvt_f16_kernel<<<blocks(NH), 256, 0, stream>>>(fB, b16, (int)NH);
        gemm(a16, b16, Hc, gateT, gate_b, m, 2 * Hc);   // split-K: a16 holds hA f16
        gate_finalize_kernel<<<blocks(NH), 256, 0, stream>>>(h, hA, fB, m, (int)NH);
        // pooling faithful to the source's negative indexing:
        // gf = pool(all_x[-1]) + pool(all_x[0]) + pool(all_x[1]) = pool(x0) + 2*pool(x1)
        pool_kernel<<<Gc, Hc, 0, stream>>>(h, out, (i == 0) ? 1.0f : 2.0f, i > 0 ? 1 : 0);
    }
}
